// GroupedQueryAttention_11106785427955
// MI455X (gfx1250) — matrix operand order repaired and checked by reading, not compiled, not run on hardware
//
#include <hip/hip_runtime.h>
#include <stdint.h>

#define N_HEADC 16
#define N_KVC   4
#define N_EMBDC 2048
#define HEAD_D  128
#define B_C     2
#define S_C     2048
#define M_TOK   (B_C * S_C)            // 4096
#define SCALE_F 0.08838834764831845f   // 128^-0.5
#define LOG2E_F 1.4426950408889634f

typedef __attribute__((ext_vector_type(16))) __bf16 bf16x16;
typedef __attribute__((ext_vector_type(8)))  __bf16 bf16x8;
typedef __attribute__((ext_vector_type(8)))  float  v8f;
typedef __attribute__((ext_vector_type(4)))  unsigned int u32x4;
typedef __attribute__((ext_vector_type(8)))  unsigned int u32x8;

static __device__ __forceinline__ unsigned short f2bf(float f) {
    unsigned int u = __float_as_uint(f);
    u += 0x7FFFu + ((u >> 16) & 1u);   // round-to-nearest-even
    return (unsigned short)(u >> 16);
}

// A-fragment (16x32 bf16, M x K): lane m = lane&15, half h = lane>>4.
static __device__ __forceinline__ bf16x16 load_fragA(const unsigned short* row, int k, int h) {
    bf16x8 lo = *reinterpret_cast<const bf16x8*>(row + k + 8 * h);
    bf16x8 hi = *reinterpret_cast<const bf16x8*>(row + k + 16 + 8 * h);
    return __builtin_shufflevector(lo, hi, 0,1,2,3,4,5,6,7,8,9,10,11,12,13,14,15);
}

// B-fragment (32x16 bf16, K x N): lane n = lane&15, half h = lane>>4.
static __device__ __forceinline__ bf16x16 load_fragB(const unsigned short* rowT, int k, int h) {
    bf16x8 lo = *reinterpret_cast<const bf16x8*>(rowT + k + 8 * h);
    bf16x8 hi = *reinterpret_cast<const bf16x8*>(rowT + k + 16 + 8 * h);
    return __builtin_shufflevector(lo, hi, 0,1,2,3,4,5,6,7,8,9,10,11,12,13,14,15);
}

static __device__ __forceinline__ v8f wmma_bf16(bf16x16 a, bf16x16 b, v8f c) {
    return __builtin_amdgcn_wmma_f32_16x16x32_bf16(false, a, false, b, (short)0, c, false, false);
}

// ---- Tensor Data Mover: 2-D tile (tile_k elems contiguous x tile_rows rows,
// row stride = stride_elems, element size 2B) from global -> LDS.
static __device__ __forceinline__ void tdm_load_2d(unsigned lds_addr, const void* gptr,
                                                   unsigned tile_k, unsigned tile_rows,
                                                   unsigned stride_elems) {
    unsigned long long ga = (unsigned long long)(uintptr_t)gptr;
    u32x4 g0;
    g0[0] = 1u;                                    // count=1 (valid user D#)
    g0[1] = lds_addr;                              // LDS byte address
    g0[2] = (unsigned)ga;                          // global_addr[31:0]
    g0[3] = (unsigned)(ga >> 32) | (2u << 30);     // global_addr[56:32] | type=2
    u32x8 g1;
    g1[0] = 0x00010000u;                           // data_size=1 (2 bytes)
    g1[1] = 2048u << 16;                           // tensor_dim0 (lo16)
    g1[2] = 4096u << 16;                           // tensor_dim1 (lo16)
    g1[3] = (tile_k & 0xFFFFu) << 16;              // tile_dim0
    g1[4] = tile_rows & 0xFFFFu;                   // tile_dim1 (tile_dim2=0)
    g1[5] = stride_elems;                          // tensor_dim0_stride[31:0]
    g1[6] = 0u;
    g1[7] = 0u;
    asm volatile("tensor_load_to_lds %0, %1" :: "s"(g0), "s"(g1) : "memory");
}

// ---------------------------------------------------------------- convert x
__global__ void __launch_bounds__(256) k_cvt_x(const float* __restrict__ x,
                                               unsigned short* __restrict__ xb) {
    int i = (blockIdx.x * 256 + threadIdx.x) * 4;
    float4 f = *reinterpret_cast<const float4*>(x + i);
    ushort4 o;
    o.x = f2bf(f.x); o.y = f2bf(f.y); o.z = f2bf(f.z); o.w = f2bf(f.w);
    *reinterpret_cast<ushort4*>(xb + i) = o;
}

// ------------------------------------------- transpose W[K][N] -> Wt[N][K] bf16
__global__ void __launch_bounds__(256) k_transpose_w(const float* __restrict__ W,
                                                     unsigned short* __restrict__ Wt,
                                                     int K, int N) {
    __shared__ unsigned short tile[32][33];
    const int n0 = blockIdx.x * 32, k0 = blockIdx.y * 32;
    const int j = threadIdx.x & 31, i0 = threadIdx.x >> 5;
#pragma unroll
    for (int kk = 0; kk < 4; ++kk) {
        int i = i0 + 8 * kk;
        tile[i][j] = f2bf(W[(size_t)(k0 + i) * N + n0 + j]);
    }
    __syncthreads();
    const int i2 = threadIdx.x & 31, j0 = threadIdx.x >> 5;
#pragma unroll
    for (int kk = 0; kk < 4; ++kk) {
        int j2 = j0 + 8 * kk;
        Wt[(size_t)(n0 + j2) * K + k0 + i2] = tile[i2][j2];
    }
}

// --------------------------------------------------------- fused QKV GEMM
// Block: 128 rows x 128 cols; waves 4(m) x 2(n), each 32x64.  Weight tiles
// (128 rows x 128 k) staged via TDM into a 64 KB LDS double buffer; each wave
// DMAs its 16-row slice.  Chunk loop kept un-unrolled so accumulators stay in
// fixed registers (avoids v_mov_b64 churn after each WMMA).
__global__ void __launch_bounds__(256) k_qkv_gemm(const unsigned short* __restrict__ xb,
                                                  const unsigned short* __restrict__ Wcat,
                                                  unsigned short* __restrict__ Qp,
                                                  unsigned short* __restrict__ Kp,
                                                  unsigned short* __restrict__ Vtp) {
    __shared__ __align__(16) unsigned short ldsB[2 * 128 * 128];   // 64 KB
    const int lane = threadIdx.x & 31;
    const int w = __builtin_amdgcn_readfirstlane(threadIdx.x >> 5);
    const int bm = blockIdx.x / 24, bn = blockIdx.x % 24;           // 32 x 24 blocks
    const int wm = w >> 1, wn = w & 1;
    const int rbase = bm * 128 + wm * 32;
    const int cb = bn * 128 + wn * 64;
    const int l16 = lane & 15, h = lane >> 4;
    const unsigned ldsbase = (unsigned)(uintptr_t)&ldsB[0];

    v8f acc[2][4];
#pragma unroll
    for (int a = 0; a < 2; ++a)
#pragma unroll
        for (int b = 0; b < 4; ++b) acc[a][b] = {};

    const unsigned short* arow0 = xb + (size_t)(rbase + l16) * 2048;
    const unsigned short* arow1 = xb + (size_t)(rbase + 16 + l16) * 2048;
    const unsigned short* gsrc = Wcat + (size_t)(bn * 128 + w * 16) * 2048; // wave slice

    const int NCH = 2048 / 128;                                     // 16 chunks
    tdm_load_2d(ldsbase + (unsigned)w * 4096u, gsrc, 128, 16, 2048);
#pragma clang loop unroll(disable)
    for (int ch = 0; ch < NCH; ++ch) {
        const int kc = ch * 128;
        if (ch + 1 < NCH) {   // issue next chunk into other buffer, wait for chunk ch
            tdm_load_2d(ldsbase + (unsigned)(((ch + 1) & 1) * 32768 + w * 4096),
                        gsrc + kc + 128, 128, 16, 2048);
            __builtin_amdgcn_s_wait_tensorcnt(1);
        } else {
            __builtin_amdgcn_s_wait_tensorcnt(0);
        }
        __syncthreads();      // chunk ch resident for all waves
        const unsigned short* bbase = ldsB + (ch & 1) * 16384;
#pragma unroll
        for (int k2 = 0; k2 < 4; ++k2) {
            const int k = kc + k2 * 32;
            __builtin_prefetch(arow0 + k + 256, 0, 3);
            __builtin_prefetch(arow1 + k + 256, 0, 3);
            bf16x16 a0 = load_fragA(arow0, k, h);
            bf16x16 a1 = load_fragA(arow1, k, h);
#pragma unroll
            for (int nt = 0; nt < 4; ++nt) {
                const unsigned short* br = bbase + (size_t)(wn * 64 + nt * 16 + l16) * 128;
                bf16x16 bf = load_fragB(br, k2 * 32, h);
                acc[0][nt] = wmma_bf16(a0, bf, acc[0][nt]);
                acc[1][nt] = wmma_bf16(a1, bf, acc[1][nt]);
            }
        }
        __syncthreads();      // reads done before TDM overwrites this buffer
    }

#pragma unroll
    for (int rh = 0; rh < 2; ++rh) {
        const int rowbase = rbase + rh * 16 + 8 * h;
#pragma unroll
        for (int nt = 0; nt < 4; ++nt) {
            const int col16 = cb + nt * 16;           // uniform per wave
            const int n = col16 + l16;
            v8f a = acc[rh][nt];
            if (col16 < 2048) {                       // ---- Q (pre-scaled)
#pragma unroll
                for (int r = 0; r < 8; ++r)
                    Qp[(size_t)(rowbase + r) * 2048 + n] = f2bf(a[r] * SCALE_F);
            } else if (col16 < 2560) {                // ---- K
                const int c = n - 2048;
#pragma unroll
                for (int r = 0; r < 8; ++r)
                    Kp[(size_t)(rowbase + r) * 512 + c] = f2bf(a[r]);
            } else {                                  // ---- V, stored transposed [d][s]
                const int v = n - 2560;
                const int kv = v >> 7, d = v & 127;
                const int b = rowbase >> 11, s = rowbase & 2047;
                __align__(16) unsigned short tmp[8];
#pragma unroll
                for (int r = 0; r < 8; ++r) tmp[r] = f2bf(a[r]);
                size_t idx = ((size_t)((b * 4 + kv) * 128 + d)) * 2048 + s;
                *reinterpret_cast<uint4*>(Vtp + idx) = *reinterpret_cast<const uint4*>(tmp);
            }
        }
    }
}

// ------------------------------------------------------ flash attention
__global__ void __launch_bounds__(256) k_attn(const unsigned short* __restrict__ Qp,
                                              const unsigned short* __restrict__ Kp,
                                              const unsigned short* __restrict__ Vtp,
                                              unsigned short* __restrict__ Op) {
    __shared__ __align__(16) unsigned short pbuf[8 * 16 * 32];
    const int lane = threadIdx.x & 31;
    const int wave = threadIdx.x >> 5;
    const int gw = blockIdx.x * 8 + wave;   // 0..4095
    const int t = gw & 127;                 // q tile index
    const int head = gw >> 7;               // 0..31
    const int b = head >> 4, hh = head & 15;
    const int kv = hh >> 2;
    const int q0 = t * 16;
    const int l16 = lane & 15, hlf = lane >> 4;
    unsigned short* pw = pbuf + wave * 512;

    const unsigned short* qrow = Qp + (size_t)(b * S_C + q0 + l16) * 2048 + hh * 128;
    bf16x16 qf[4];
#pragma unroll
    for (int c = 0; c < 4; ++c) qf[c] = load_fragA(qrow, c * 32, hlf);

    v8f O[8];
#pragma unroll
    for (int c = 0; c < 8; ++c) O[c] = {};
    float mrow[8], lrow[8];
#pragma unroll
    for (int r = 0; r < 8; ++r) { mrow[r] = -3.0e38f; lrow[r] = 0.0f; }

    const int nb = (t + 2) >> 1;
    for (int blk = 0; blk < nb; ++blk) {
        const int sbase = blk * 32;
        v8f st[2];
#pragma unroll
        for (int tt = 0; tt < 2; ++tt) {
            v8f sc = {};
            const unsigned short* krow =
                Kp + (size_t)(b * S_C + sbase + tt * 16 + l16) * 512 + kv * 128;
#pragma unroll
            for (int c = 0; c < 4; ++c)
                sc = wmma_bf16(qf[c], load_fragB(krow, c * 32, hlf), sc);
            st[tt] = sc;
        }
        float p0[8], p1[8], alpha[8];
#pragma unroll
        for (int r = 0; r < 8; ++r) {
            const int q = q0 + r + 8 * hlf;
            const int k0i = sbase + l16, k1i = sbase + 16 + l16;
            float s0 = (k0i > q) ? -3.0e38f : st[0][r];
            float s1 = (k1i > q) ? -3.0e38f : st[1][r];
            float mx = fmaxf(s0, s1);
#pragma unroll
            for (int mk = 1; mk < 16; mk <<= 1) mx = fmaxf(mx, __shfl_xor(mx, mk, 32));
            const float mnew = fmaxf(mrow[r], mx);
            alpha[r] = exp2f((mrow[r] - mnew) * LOG2E_F);
            p0[r] = exp2f((s0 - mnew) * LOG2E_F);
            p1[r] = exp2f((s1 - mnew) * LOG2E_F);
            float rs = p0[r] + p1[r];
#pragma unroll
            for (int mk = 1; mk < 16; mk <<= 1) rs += __shfl_xor(rs, mk, 32);
            lrow[r] = lrow[r] * alpha[r] + rs;
            mrow[r] = mnew;
        }
#pragma unroll
        for (int r = 0; r < 8; ++r) {
            const int m = r + 8 * hlf;
            pw[m * 32 + l16] = f2bf(p0[r]);
            pw[m * 32 + 16 + l16] = f2bf(p1[r]);
        }
        asm volatile("s_wait_dscnt 0" ::: "memory");
        bf16x16 pf = load_fragA(pw + l16 * 32, 0, hlf);
#pragma unroll
        for (int c = 0; c < 8; ++c) {
#pragma unroll
            for (int r = 0; r < 8; ++r) O[c][r] *= alpha[r];
            const unsigned short* vrow =
                Vtp + (size_t)((b * 4 + kv) * 128 + c * 16 + l16) * 2048 + sbase;
            O[c] = wmma_bf16(pf, load_fragB(vrow, 0, hlf), O[c]);
        }
    }
#pragma unroll
    for (int r = 0; r < 8; ++r) lrow[r] = 1.0f / lrow[r];
#pragma unroll
    for (int c = 0; c < 8; ++c)
#pragma unroll
        for (int r = 0; r < 8; ++r) {
            const int m = r + 8 * hlf;
            Op[(size_t)(b * S_C + q0 + m) * 2048 + hh * 128 + c * 16 + l16] =
                f2bf(O[c][r] * lrow[r]);
        }
}

// --------------------------------------------------------- output projection
__global__ void __launch_bounds__(256) k_oproj(const unsigned short* __restrict__ Ab,
                                               const unsigned short* __restrict__ WoT,
                                               float* __restrict__ out) {
    __shared__ __align__(16) unsigned short ldsB[2 * 128 * 128];   // 64 KB
    const int lane = threadIdx.x & 31;
    const int w = __builtin_amdgcn_readfirstlane(threadIdx.x >> 5);
    const int bm = blockIdx.x >> 4, bn = blockIdx.x & 15;           // 32 x 16 blocks
    const int wm = w >> 1, wn = w & 1;
    const int rbase = bm * 128 + wm * 32;
    const int cb = bn * 128 + wn * 64;
    const int l16 = lane & 15, h = lane >> 4;
    const unsigned ldsbase = (unsigned)(uintptr_t)&ldsB[0];

    v8f acc[2][4];
#pragma unroll
    for (int a = 0; a < 2; ++a)
#pragma unroll
        for (int b = 0; b < 4; ++b) acc[a][b] = {};

    const unsigned short* arow0 = Ab + (size_t)(rbase + l16) * 2048;
    const unsigned short* arow1 = Ab + (size_t)(rbase + 16 + l16) * 2048;
    const unsigned short* gsrc = WoT + (size_t)(bn * 128 + w * 16) * 2048;

    const int NCH = 2048 / 128;
    tdm_load_2d(ldsbase + (unsigned)w * 4096u, gsrc, 128, 16, 2048);
#pragma clang loop unroll(disable)
    for (int ch = 0; ch < NCH; ++ch) {
        const int kc = ch * 128;
        if (ch + 1 < NCH) {
            tdm_load_2d(ldsbase + (unsigned)(((ch + 1) & 1) * 32768 + w * 4096),
                        gsrc + kc + 128, 128, 16, 2048);
            __builtin_amdgcn_s_wait_tensorcnt(1);
        } else {
            __builtin_amdgcn_s_wait_tensorcnt(0);
        }
        __syncthreads();
        const unsigned short* bbase = ldsB + (ch & 1) * 16384;
#pragma unroll
        for (int k2 = 0; k2 < 4; ++k2) {
            const int k = kc + k2 * 32;
            __builtin_prefetch(arow0 + k + 256, 0, 3);
            __builtin_prefetch(arow1 + k + 256, 0, 3);
            bf16x16 a0 = load_fragA(arow0, k, h);
            bf16x16 a1 = load_fragA(arow1, k, h);
#pragma unroll
            for (int nt = 0; nt < 4; ++nt) {
                const unsigned short* br = bbase + (size_t)(wn * 64 + nt * 16 + l16) * 128;
                bf16x16 bf = load_fragB(br, k2 * 32, h);
                acc[0][nt] = wmma_bf16(a0, bf, acc[0][nt]);
                acc[1][nt] = wmma_bf16(a1, bf, acc[1][nt]);
            }
        }
        __syncthreads();
    }
#pragma unroll
    for (int rh = 0; rh < 2; ++rh) {
        const int rowbase = rbase + rh * 16 + 8 * h;
#pragma unroll
        for (int nt = 0; nt < 4; ++nt) {
            const int col = cb + nt * 16 + l16;
            v8f a = acc[rh][nt];
#pragma unroll
            for (int r = 0; r < 8; ++r)
                out[(size_t)(rowbase + r) * 2048 + col] = a[r];
        }
    }
}

// --------------------------------------------------------------- launcher
extern "C" void kernel_launch(void* const* d_in, const int* in_sizes, int n_in,
                              void* d_out, int out_size, void* d_ws, size_t ws_size,
                              hipStream_t stream) {
    (void)in_sizes; (void)n_in; (void)out_size; (void)ws_size;
    const float* x  = (const float*)d_in[0];
    const float* Wq = (const float*)d_in[1];
    const float* Wk = (const float*)d_in[2];
    const float* Wv = (const float*)d_in[3];
    const float* Wo = (const float*)d_in[4];
    float* out = (float*)d_out;

    char* ws = (char*)d_ws;
    unsigned short* xb   = (unsigned short*)(ws);               // 16 MB
    unsigned short* Wcat = (unsigned short*)(ws + (16u << 20)); // 12 MB (3072x2048)
    unsigned short* WoT  = (unsigned short*)(ws + (28u << 20)); //  8 MB
    unsigned short* Qp   = (unsigned short*)(ws + (36u << 20)); // 16 MB
    unsigned short* Kp   = (unsigned short*)(ws + (52u << 20)); //  4 MB
    unsigned short* Vtp  = (unsigned short*)(ws + (56u << 20)); //  4 MB
    unsigned short* Ab   = (unsigned short*)(ws + (60u << 20)); // 16 MB  (total 76 MB)

    k_cvt_x<<<8192, 256, 0, stream>>>(x, xb);
    k_transpose_w<<<dim3(64, 64), 256, 0, stream>>>(Wq, Wcat, 2048, 2048);
    k_transpose_w<<<dim3(16, 64), 256, 0, stream>>>(Wk, Wcat + (size_t)2048 * 2048, 2048, 512);
    k_transpose_w<<<dim3(16, 64), 256, 0, stream>>>(Wv, Wcat + (size_t)2560 * 2048, 2048, 512);
    k_transpose_w<<<dim3(64, 64), 256, 0, stream>>>(Wo, WoT, 2048, 2048);

    k_qkv_gemm<<<768, 256, 0, stream>>>(xb, Wcat, Qp, Kp, Vtp);  // 32x24 blocks
    k_attn<<<512, 256, 0, stream>>>(Qp, Kp, Vtp, Ab);            // 4096 wave tiles
    k_oproj<<<512, 256, 0, stream>>>(Ab, WoT, out);              // 32x16 blocks
}
